// RLMambaBlock_2705829396663
// MI455X (gfx1250) — matrix operand order repaired and checked by reading, not compiled, not run on hardware
//
#include <hip/hip_runtime.h>
#include <hip/hip_bf16.h>
#include <math.h>

typedef unsigned short u16;
typedef __attribute__((ext_vector_type(16))) __bf16 v16bf;
typedef __attribute__((ext_vector_type(8)))  float  v8f;
typedef __attribute__((ext_vector_type(4)))  unsigned int u32x4;

namespace {
constexpr int BB  = 2;
constexpr int LL  = 1024;
constexpr int DD  = 512;
constexpr int DI  = 1024;   // d_inner
constexpr int NS  = 16;     // d_state
constexpr int KC  = 4;      // d_conv
constexpr int TOK = BB * LL;  // 2048 tokens
}

// ---------------------------------------------------------------- helpers
__device__ __forceinline__ u16 f2bf(float f) {
  union { float f; unsigned int u; } c; c.f = f;
  unsigned int r = c.u + 0x7FFFu + ((c.u >> 16) & 1u);   // round-nearest-even
  return (u16)(r >> 16);
}
__device__ __forceinline__ float siluf(float x)     { return x / (1.f + __expf(-x)); }
__device__ __forceinline__ float softplusf(float x) { return (x > 20.f) ? x : log1pf(__expf(x)); }
__device__ __forceinline__ float geluf(float x)     { return 0.5f * x * (1.f + erff(x * 0.70710678118654752f)); }

// ---------------------------------------------------------------- f32 -> bf16
__global__ __launch_bounds__(256) void cvt_bf16_kernel(const float* __restrict__ in,
                                                       u16* __restrict__ out, int n) {
  int i = blockIdx.x * 256 + threadIdx.x;
  if (i < n) out[i] = f2bf(in[i]);
}

// ---------------------------------------------------------------- WMMA GEMM
// C[m,n] = sum_k A[m,k] * W[n,k]   (A: [M,K] bf16, W: [N,K] bf16 => B = W^T)
// Wave tile 32x32 (2x2 WMMA frags, 4 independent accumulator chains per K-step).
// Block = 4 waves (2x2) -> 64x64 C tile. Fragment layouts per CDNA5 ISA 7.12.2.
union FragBF { v16bf v; u32x4 q[2]; };

// epilogue modes
//  0: store f32
//  1: store f32 + bf16 copy
//  2: +bias, softplus, store f32
//  3: +bias, gelu(exact), store bf16 only
//  4: store f32 = aux - C                (res1 = x - mamba_out)
//  5: store f32 = aux - (C + bias)       (res2 = r - ff)
__device__ __forceinline__ void gemm_epilogue(
    float* __restrict__ C, int ldc, int mode,
    const float* __restrict__ bias, const float* __restrict__ aux,
    u16* __restrict__ outb, int ldob,
    int baseM, int col, int hi, v8f acc)
{
#pragma unroll
  for (int r = 0; r < 8; ++r) {
    const int row = baseM + r + hi * 8;
    float v = acc[r];
    switch (mode) {
      case 0:
        C[(size_t)row * ldc + col] = v; break;
      case 1:
        C[(size_t)row * ldc + col] = v;
        outb[(size_t)row * ldob + col] = f2bf(v); break;
      case 2:
        v = softplusf(v + bias[col]);
        C[(size_t)row * ldc + col] = v; break;
      case 3:
        v = geluf(v + bias[col]);
        outb[(size_t)row * ldob + col] = f2bf(v); break;
      case 4:
        C[(size_t)row * ldc + col] = aux[(size_t)row * ldc + col] - v; break;
      case 5:
        C[(size_t)row * ldc + col] = aux[(size_t)row * ldc + col] - (v + bias[col]); break;
    }
  }
}

__global__ __launch_bounds__(128) void wmma_gemm_kernel(
    const u16* __restrict__ A, int lda,
    const u16* __restrict__ W, int ldw,
    float* __restrict__ C, int ldc,
    int K, int mode,
    const float* __restrict__ bias,
    const float* __restrict__ aux,
    u16* __restrict__ outb, int ldob)
{
  const int lane  = threadIdx.x & 31;
  const int wave  = threadIdx.x >> 5;
  const int tileM = blockIdx.y * 64 + (wave >> 1) * 32;   // wave owns 32x32 of C
  const int tileN = blockIdx.x * 64 + (wave & 1) * 32;
  const int mn = lane & 15;      // A row within frag / B col within frag
  const int hi = lane >> 4;      // lane half selects K sub-block

  const u16* Arow0 = A + (size_t)(tileM + mn) * lda + hi * 8;   // K{koff..+7}, K{koff+16..+23}
  const u16* Arow1 = Arow0 + (size_t)16 * lda;
  const u16* Wrow0 = W + (size_t)(tileN + mn) * ldw + hi * 8;   // K{koff..+7}, K{koff+16..+23}
  const u16* Wrow1 = Wrow0 + (size_t)16 * ldw;

  v8f acc00 = {}, acc01 = {}, acc10 = {}, acc11 = {};
  for (int k0 = 0; k0 < K; k0 += 32) {
    if (k0 + 32 < K) {                         // speculative prefetch of next slab
      __builtin_prefetch(Arow0 + k0 + 32, 0, 1);
      __builtin_prefetch(Arow1 + k0 + 32, 0, 1);
      __builtin_prefetch(Wrow0 + k0 + 32, 0, 1);
      __builtin_prefetch(Wrow1 + k0 + 32, 0, 1);
    }
    FragBF a0, a1, b0, b1;
    a0.q[0] = *reinterpret_cast<const u32x4*>(Arow0 + k0);
    a0.q[1] = *reinterpret_cast<const u32x4*>(Arow0 + k0 + 16);
    a1.q[0] = *reinterpret_cast<const u32x4*>(Arow1 + k0);
    a1.q[1] = *reinterpret_cast<const u32x4*>(Arow1 + k0 + 16);
    b0.q[0] = *reinterpret_cast<const u32x4*>(Wrow0 + k0);
    b0.q[1] = *reinterpret_cast<const u32x4*>(Wrow0 + k0 + 16);
    b1.q[0] = *reinterpret_cast<const u32x4*>(Wrow1 + k0);
    b1.q[1] = *reinterpret_cast<const u32x4*>(Wrow1 + k0 + 16);
    acc00 = __builtin_amdgcn_wmma_f32_16x16x32_bf16(false, a0.v, false, b0.v, (short)0, acc00, false, false);
    acc01 = __builtin_amdgcn_wmma_f32_16x16x32_bf16(false, a0.v, false, b1.v, (short)0, acc01, false, false);
    acc10 = __builtin_amdgcn_wmma_f32_16x16x32_bf16(false, a1.v, false, b0.v, (short)0, acc10, false, false);
    acc11 = __builtin_amdgcn_wmma_f32_16x16x32_bf16(false, a1.v, false, b1.v, (short)0, acc11, false, false);
  }

  const int col0 = tileN + mn;
  gemm_epilogue(C, ldc, mode, bias, aux, outb, ldob, tileM,      col0,      hi, acc00);
  gemm_epilogue(C, ldc, mode, bias, aux, outb, ldob, tileM,      col0 + 16, hi, acc01);
  gemm_epilogue(C, ldc, mode, bias, aux, outb, ldob, tileM + 16, col0,      hi, acc10);
  gemm_epilogue(C, ldc, mode, bias, aux, outb, ldob, tileM + 16, col0 + 16, hi, acc11);
}

// ---------------------------------------------------------------- depthwise causal conv + SiLU
// u[t,d] = silu( conv_b[d] + sum_k cw[d,k] * xz[t-3+k, d] ) within batch; xz cols 0..DI-1
__global__ __launch_bounds__(256) void conv_silu_kernel(
    const float* __restrict__ xz, const float* __restrict__ cw, const float* __restrict__ cb,
    float* __restrict__ uc, u16* __restrict__ ucb)
{
  int i = blockIdx.x * 256 + threadIdx.x;      // [0, TOK*DI)
  int d = i & (DI - 1);
  int t = i >> 10;
  int l = t & (LL - 1);
  float s = cb[d];
#pragma unroll
  for (int k = 0; k < KC; ++k) {
    int ls = l - (KC - 1) + k;
    if (ls >= 0) s += cw[d * KC + k] * xz[(size_t)(t - (KC - 1) + k) * (2 * DI) + d];
  }
  float v = siluf(s);
  uc[i]  = v;
  ucb[i] = f2bf(v);
}

// ---------------------------------------------------------------- selective scan (sequential in L)
// one thread per (b,d) row; 16 states in registers; B/C chunks staged in LDS.
// fused epilogue: yg = (y + u*D_skip) * silu(z), stored bf16 for out_proj.
__global__ __launch_bounds__(256) void scan_kernel(
    const float* __restrict__ dt, const float* __restrict__ uc,
    const float* __restrict__ xdbl,   // [TOK,64]: cols 32..47 = B, 48..63 = C
    const float* __restrict__ xz,     // z at cols DI..2DI-1
    const float* __restrict__ A_log, const float* __restrict__ D_skip,
    u16* __restrict__ yg)
{
  __shared__ float shB[16][16];
  __shared__ float shC[16][16];
  const int bid = blockIdx.x;                    // 8 blocks: b*4 + dgroup
  const int b   = bid >> 2;
  const int d   = ((bid & 3) << 8) + threadIdx.x;

  float An[NS], h[NS];
#pragma unroll
  for (int n = 0; n < NS; ++n) { An[n] = -__expf(A_log[d * NS + n]); h[n] = 0.f; }
  const float Dsk = D_skip[d];

  for (int l0 = 0; l0 < LL; l0 += 16) {
    // stage B/C for 16 timesteps: 512 floats, 2 per thread
    for (int j = threadIdx.x; j < 512; j += 256) {
      int li = j >> 5, c = j & 31;
      float v = xdbl[(size_t)(b * LL + l0 + li) * 64 + 32 + c];
      if (c < 16) shB[li][c] = v; else shC[li][c - 16] = v;
    }
    __syncthreads();
#pragma unroll 1
    for (int j = 0; j < 16; ++j) {
      const size_t t = (size_t)(b * LL + l0 + j);
      const float dtv = dt[t * DI + d];
      const float uv  = uc[t * DI + d];
      float acc = 0.f;
#pragma unroll
      for (int n = 0; n < NS; ++n) {
        float dA = __expf(dtv * An[n]);
        h[n] = dA * h[n] + dtv * shB[j][n] * uv;
        acc += h[n] * shC[j][n];
      }
      const float zv = xz[t * (2 * DI) + DI + d];
      const float y  = (acc + uv * Dsk) * siluf(zv);
      yg[t * DI + d] = f2bf(y);
    }
    __syncthreads();
  }
}

// ---------------------------------------------------------------- LayerNorm (D=512), block per token
__global__ __launch_bounds__(256) void ln1_kernel(
    const float* __restrict__ src, const float* __restrict__ g, const float* __restrict__ be,
    float* __restrict__ dst, u16* __restrict__ dstb)
{
  __shared__ float red[256];
  const int t = blockIdx.x, tid = threadIdx.x;
  const float* xr = src + (size_t)t * DD;
  float v0 = xr[tid], v1 = xr[tid + 256];
  red[tid] = v0 + v1; __syncthreads();
  for (int s = 128; s > 0; s >>= 1) { if (tid < s) red[tid] += red[tid + s]; __syncthreads(); }
  float mean = red[0] * (1.f / DD); __syncthreads();
  float d0 = v0 - mean, d1 = v1 - mean;
  red[tid] = d0 * d0 + d1 * d1; __syncthreads();
  for (int s = 128; s > 0; s >>= 1) { if (tid < s) red[tid] += red[tid + s]; __syncthreads(); }
  float inv = rsqrtf(red[0] * (1.f / DD) + 1e-5f);
  float o0 = d0 * inv * g[tid] + be[tid];
  float o1 = d1 * inv * g[tid + 256] + be[tid + 256];
  dst[(size_t)t * DD + tid]        = o0;  dstb[(size_t)t * DD + tid]        = f2bf(o0);
  dst[(size_t)t * DD + tid + 256]  = o1;  dstb[(size_t)t * DD + tid + 256]  = f2bf(o1);
}

// final: out = LN(silu(res2), g2, b2) + x
__global__ __launch_bounds__(256) void final_ln_kernel(
    const float* __restrict__ res2, const float* __restrict__ x,
    const float* __restrict__ g, const float* __restrict__ be, float* __restrict__ out)
{
  __shared__ float red[256];
  const int t = blockIdx.x, tid = threadIdx.x;
  const float* rr = res2 + (size_t)t * DD;
  float v0 = siluf(rr[tid]), v1 = siluf(rr[tid + 256]);
  red[tid] = v0 + v1; __syncthreads();
  for (int s = 128; s > 0; s >>= 1) { if (tid < s) red[tid] += red[tid + s]; __syncthreads(); }
  float mean = red[0] * (1.f / DD); __syncthreads();
  float d0 = v0 - mean, d1 = v1 - mean;
  red[tid] = d0 * d0 + d1 * d1; __syncthreads();
  for (int s = 128; s > 0; s >>= 1) { if (tid < s) red[tid] += red[tid + s]; __syncthreads(); }
  float inv = rsqrtf(red[0] * (1.f / DD) + 1e-5f);
  out[(size_t)t * DD + tid]       = d0 * inv * g[tid]       + be[tid]       + x[(size_t)t * DD + tid];
  out[(size_t)t * DD + tid + 256] = d1 * inv * g[tid + 256] + be[tid + 256] + x[(size_t)t * DD + tid + 256];
}

// ---------------------------------------------------------------- launch
extern "C" void kernel_launch(void* const* d_in, const int* in_sizes, int n_in,
                              void* d_out, int out_size, void* d_ws, size_t ws_size,
                              hipStream_t stream) {
  (void)in_sizes; (void)n_in; (void)out_size; (void)ws_size;
  const float* x      = (const float*)d_in[0];
  const float* W_in   = (const float*)d_in[1];
  const float* conv_w = (const float*)d_in[2];
  const float* conv_b = (const float*)d_in[3];
  const float* W_xproj= (const float*)d_in[4];
  const float* W_dt   = (const float*)d_in[5];
  const float* b_dt   = (const float*)d_in[6];
  const float* A_log  = (const float*)d_in[7];
  const float* D_skip = (const float*)d_in[8];
  const float* W_out  = (const float*)d_in[9];
  const float* g1     = (const float*)d_in[10];
  const float* b1     = (const float*)d_in[11];
  const float* W_ff1  = (const float*)d_in[12];
  const float* b_ff1  = (const float*)d_in[13];
  const float* W_ff2  = (const float*)d_in[14];
  const float* b_ff2  = (const float*)d_in[15];
  const float* g2     = (const float*)d_in[16];
  const float* b2     = (const float*)d_in[17];
  float* out = (float*)d_out;

  size_t off = 0;
  auto alloc = [&](size_t bytes) -> void* {
    void* p = (char*)d_ws + off;
    off += (bytes + 255) & ~(size_t)255;
    return p;
  };
  // bf16 operand buffers
  u16* xb     = (u16*)alloc((size_t)TOK * DD * 2);
  u16* Wb_in  = (u16*)alloc((size_t)2 * DI * DD * 2);
  u16* Wb_xp  = (u16*)alloc((size_t)64 * DI * 2);
  u16* Wb_dt  = (u16*)alloc((size_t)DI * 32 * 2);
  u16* Wb_out = (u16*)alloc((size_t)DD * DI * 2);
  u16* Wb_ff1 = (u16*)alloc((size_t)2 * DD * DD * 2);
  u16* Wb_ff2 = (u16*)alloc((size_t)DD * 2 * DD * 2);
  u16* ucb    = (u16*)alloc((size_t)TOK * DI * 2);
  u16* xdbl_b = (u16*)alloc((size_t)TOK * 64 * 2);
  u16* ygb    = (u16*)alloc((size_t)TOK * DI * 2);
  u16* rb     = (u16*)alloc((size_t)TOK * DD * 2);
  u16* h1b    = (u16*)alloc((size_t)TOK * DI * 2);
  // f32 buffers
  float* xz   = (float*)alloc((size_t)TOK * 2 * DI * 4);
  float* uc   = (float*)alloc((size_t)TOK * DI * 4);
  float* xdbl = (float*)alloc((size_t)TOK * 64 * 4);
  float* dtb  = (float*)alloc((size_t)TOK * DI * 4);
  float* res1 = (float*)alloc((size_t)TOK * DD * 4);
  float* rbuf = (float*)alloc((size_t)TOK * DD * 4);
  float* res2 = (float*)alloc((size_t)TOK * DD * 4);

  auto cvt = [&](const float* src, u16* dst, int n) {
    cvt_bf16_kernel<<<(n + 255) / 256, 256, 0, stream>>>(src, dst, n);
  };
  cvt(x,      xb,     TOK * DD);
  cvt(W_in,   Wb_in,  2 * DI * DD);
  cvt(W_xproj,Wb_xp,  64 * DI);
  cvt(W_dt,   Wb_dt,  DI * 32);
  cvt(W_out,  Wb_out, DD * DI);
  cvt(W_ff1,  Wb_ff1, 2 * DD * DD);
  cvt(W_ff2,  Wb_ff2, DD * 2 * DD);

  // 1) in_proj: xz = x @ W_in^T    [2048 x 2048], K=512
  wmma_gemm_kernel<<<dim3(2 * DI / 64, TOK / 64), 128, 0, stream>>>(
      xb, DD, Wb_in, DD, xz, 2 * DI, DD, 0, nullptr, nullptr, nullptr, 0);
  // 2) depthwise conv + SiLU -> uc (f32 + bf16)
  conv_silu_kernel<<<TOK * DI / 256, 256, 0, stream>>>(xz, conv_w, conv_b, uc, ucb);
  // 3) x_proj: x_dbl = u @ W_xproj^T   [2048 x 64], K=1024
  wmma_gemm_kernel<<<dim3(64 / 64, TOK / 64), 128, 0, stream>>>(
      ucb, DI, Wb_xp, DI, xdbl, 64, DI, 1, nullptr, nullptr, xdbl_b, 64);
  // 4) dt = softplus(dt_r @ W_dt^T + b_dt)   [2048 x 1024], K=32 (lda=64, first 32 cols of x_dbl)
  wmma_gemm_kernel<<<dim3(DI / 64, TOK / 64), 128, 0, stream>>>(
      xdbl_b, 64, Wb_dt, 32, dtb, DI, 32, 2, b_dt, nullptr, nullptr, 0);
  // 5) selective scan + (+u*D)*silu(z) -> ygb bf16
  scan_kernel<<<BB * (DI / 256), 256, 0, stream>>>(dtb, uc, xdbl, xz, A_log, D_skip, ygb);
  // 6) out_proj + residual: res1 = x - yg @ W_out^T   [2048 x 512], K=1024
  wmma_gemm_kernel<<<dim3(DD / 64, TOK / 64), 128, 0, stream>>>(
      ygb, DI, Wb_out, DI, res1, DD, DI, 4, nullptr, x, nullptr, 0);
  // 7) r = LN(res1, g1, b1)  (f32 + bf16)
  ln1_kernel<<<TOK, 256, 0, stream>>>(res1, g1, b1, rbuf, rb);
  // 8) ff1: h1 = gelu(r @ W_ff1^T + b_ff1)  [2048 x 1024], K=512 -> bf16
  wmma_gemm_kernel<<<dim3(2 * DD / 64, TOK / 64), 128, 0, stream>>>(
      rb, DD, Wb_ff1, DD, nullptr, 2 * DD, DD, 3, b_ff1, nullptr, h1b, 2 * DD);
  // 9) ff2 + residual: res2 = r - (h1 @ W_ff2^T + b_ff2)  [2048 x 512], K=1024
  wmma_gemm_kernel<<<dim3(DD / 64, TOK / 64), 128, 0, stream>>>(
      h1b, 2 * DD, Wb_ff2, 2 * DD, res2, DD, 2 * DD, 5, b_ff2, rbuf, nullptr, 0);
  // 10) out = LN(silu(res2), g2, b2) + x
  final_ln_kernel<<<TOK, 256, 0, stream>>>(res2, x, g2, b2, out);
}
